// MultiHeadsLatentAttention_21019569946913
// MI455X (gfx1250) — matrix operand order repaired and checked by reading, not compiled, not run on hardware
//
#include <hip/hip_runtime.h>

typedef __attribute__((ext_vector_type(16))) __bf16 v16bf;
typedef __attribute__((ext_vector_type(8)))  float  v8f;
typedef __attribute__((ext_vector_type(4)))  int    v4i;

#define AS_GLOBAL __attribute__((address_space(1)))
#define AS_LDS    __attribute__((address_space(3)))

#if defined(__has_builtin)
# if __has_builtin(__builtin_amdgcn_global_load_async_to_lds_b128)
#  define HAVE_ASYNC_LDS 1
# endif
# if __has_builtin(__builtin_amdgcn_s_wait_asynccnt)
#  define HAVE_WAIT_ASYNC 1
# endif
#endif

union Frag {
  v16bf v;
  float4 f[2];
};
union Chunk8 {
  float4 f;
  __bf16 h[8];
};

__device__ __forceinline__ v8f wmma_bf16(v16bf a, v16bf b, v8f c) {
  // D = A(16x32 bf16) * B(32x16 bf16) + C(16x16 f32)
  return __builtin_amdgcn_wmma_f32_16x16x32_bf16(false, a, false, b, (short)0, c, false, false);
}

#define DS_WAIT() asm volatile("s_wait_dscnt 0" ::: "memory")

// 16B global->LDS copy. Prefers the CDNA5 async-to-LDS path (ASYNCcnt-tracked,
// no VGPR round trip); falls back to load+ds_store if the builtin is absent.
__device__ __forceinline__ void async_copy_b128(__bf16* lds_dst, const __bf16* gsrc) {
#ifdef HAVE_ASYNC_LDS
  __builtin_amdgcn_global_load_async_to_lds_b128(
      (AS_GLOBAL v4i*)gsrc, (AS_LDS v4i*)lds_dst, 0, 0);
#else
  *(float4*)lds_dst = *(const float4*)gsrc;
#endif
}

__device__ __forceinline__ void wait_async_lds() {
#ifdef HAVE_ASYNC_LDS
# ifdef HAVE_WAIT_ASYNC
  __builtin_amdgcn_s_wait_asynccnt(0);
# else
  asm volatile("s_wait_asynccnt 0" ::: "memory");
# endif
#endif
}

__global__ void cvt_f32_bf16_kernel(const float* __restrict__ src, __bf16* __restrict__ dst, int n) {
  int i = blockIdx.x * blockDim.x + threadIdx.x;
  if (i < n) dst[i] = (__bf16)src[i];
}

// C = A(MxK) @ B(KxN), bf16 operands, f32 accumulation via v_wmma_f32_16x16x32_bf16.
// Block tile 128x128, 8 waves, each wave 32x64 (2x4 WMMA tiles). K-step 32.
// A tile staged via async-to-LDS (row-major, stride 40 bf16); B tile staged
// transposed in registers (Bt[col][k], stride 40) so B fragments are two
// contiguous ds_load_b128 per WMMA.
// mode 0: store bf16 row-major (ldc = N)
// mode 1: store bf16 transposed  C[col*ldc + row] (ldc = M)
// mode 2: store f32 row-major (ldc = N)
__global__ __launch_bounds__(256)
void gemm_bf16_wmma(const __bf16* __restrict__ A, const __bf16* __restrict__ B,
                    void* __restrict__ C, int M, int N, int K,
                    int lda, int ldb, int ldc, int mode) {
  __shared__ __bf16 As[128 * 40];
  __shared__ __bf16 Bt[128 * 40];
  const int tid  = threadIdx.x;
  const int row0 = blockIdx.y * 128;
  const int col0 = blockIdx.x * 128;
  const int wave = tid >> 5, lane = tid & 31;
  const int wr = wave >> 1, wc = wave & 1;
  const int m16 = lane & 15, lh = lane >> 4;

  v8f acc[2][4];
  for (int rt = 0; rt < 2; ++rt)
    for (int ct = 0; ct < 4; ++ct)
      for (int r = 0; r < 8; ++r) acc[rt][ct][r] = 0.f;

  for (int k0 = 0; k0 < K; k0 += 32) {
    // Stage A tile: 128 rows x 32 k, two async 16B copies per thread.
    for (int i = 0; i < 2; ++i) {
      int c = tid + i * 256;
      int r = c >> 2, ch = c & 3;
      async_copy_b128(&As[r * 40 + ch * 8],
                      A + (size_t)(row0 + r) * lda + k0 + ch * 8);
    }
    // Stage B tile transposed in registers: Bt[col][k].
    for (int i = 0; i < 2; ++i) {
      int c = tid + i * 256;
      int kr = c >> 4, ch = c & 15;
      Chunk8 d;
      d.f = *(const float4*)(B + (size_t)(k0 + kr) * ldb + col0 + ch * 8);
      for (int e = 0; e < 8; ++e) Bt[(ch * 8 + e) * 40 + kr] = d.h[e];
    }
    // Prefetch next K-step tiles into cache while this tile computes.
    if (k0 + 32 < K) {
      __builtin_prefetch(A + (size_t)(row0 + (tid >> 2)) * lda + k0 + 32, 0, 3);
      __builtin_prefetch(B + (size_t)(k0 + 32 + (tid >> 4)) * ldb + col0 + (tid & 15) * 8, 0, 3);
    }
    wait_async_lds();
    __syncthreads();

    Frag af[2], bfr[4];
    // A fragment: lane m16 = row, lane-half selects k-base 0/8; VGPRs 0-3 hold k..k+7,
    // VGPRs 4-7 hold k+16..k+23 -> two contiguous 16B LDS loads.
    for (int rt = 0; rt < 2; ++rt) {
      const __bf16* p = &As[(wr * 32 + rt * 16 + m16) * 40 + lh * 8];
      af[rt].f[0] = *(const float4*)(p);
      af[rt].f[1] = *(const float4*)(p + 16);
    }
    // B fragment: lane m16 = column, lane-half selects k-base 0/8; VGPRs 0-3 hold k..k+7,
    // VGPRs 4-7 hold k+16..k+23 -> two contiguous 16B LDS loads.
    for (int ct = 0; ct < 4; ++ct) {
      const __bf16* p = &Bt[(wc * 64 + ct * 16 + m16) * 40 + lh * 8];
      bfr[ct].f[0] = *(const float4*)(p);
      bfr[ct].f[1] = *(const float4*)(p + 16);
    }
    for (int rt = 0; rt < 2; ++rt)
      for (int ct = 0; ct < 4; ++ct)
        acc[rt][ct] = wmma_bf16(af[rt].v, bfr[ct].v, acc[rt][ct]);
    __syncthreads();
  }

  // C layout: lane m16 = column, lane-half selects rows 0-7 / 8-15, VGPR r = row.
  // All-int32 offset math inside the loops; one 64-bit base per 16x16 tile.
  if (mode == 0) {
    __bf16* Cb = (__bf16*)C;
    for (int rt = 0; rt < 2; ++rt)
      for (int ct = 0; ct < 4; ++ct) {
        __bf16* p = Cb + (size_t)((row0 + wr * 32 + rt * 16 + lh * 8) * ldc
                                  + (col0 + wc * 64 + ct * 16 + m16));
        for (int r = 0; r < 8; ++r) p[r * ldc] = (__bf16)acc[rt][ct][r];
      }
  } else if (mode == 1) {
    __bf16* Cb = (__bf16*)C;
    for (int rt = 0; rt < 2; ++rt)
      for (int ct = 0; ct < 4; ++ct) {
        __bf16* p = Cb + (size_t)((col0 + wc * 64 + ct * 16 + m16) * ldc
                                  + (row0 + wr * 32 + rt * 16 + lh * 8));
        for (int r = 0; r < 8; ++r) p[r] = (__bf16)acc[rt][ct][r];
      }
  } else {
    float* Cf = (float*)C;
    for (int rt = 0; rt < 2; ++rt)
      for (int ct = 0; ct < 4; ++ct) {
        float* p = Cf + (size_t)((row0 + wr * 32 + rt * 16 + lh * 8) * ldc
                                 + (col0 + wc * 64 + ct * 16 + m16));
        for (int r = 0; r < 8; ++r) p[r * ldc] = acc[rt][ct][r];
      }
  }
}

// Causal flash attention over 16 heads, head_dim 64, T=4096, layout (t, h*64+d).
// One wave handles a 16-query block of one head. V is pre-transposed: Vt[(h*64+d)*4096 + t].
__global__ __launch_bounds__(128)
void mla_flash_attn(const __bf16* __restrict__ Q, const __bf16* __restrict__ Km,
                    const __bf16* __restrict__ Vt, __bf16* __restrict__ Ctx) {
  __shared__ __bf16 Pb[4 * 16 * 40];  // per-wave 16x32 P tile, stride 40
  const int h    = blockIdx.x;
  const int wave = threadIdx.x >> 5, lane = threadIdx.x & 31;
  const int q0   = (blockIdx.y * 4 + wave) * 16;
  const int m16  = lane & 15, lh = lane >> 4;
  __bf16* Pw = &Pb[wave * 640];

  // Q tile: two 16x32 A fragments (head_dim 64).
  Frag qa[2];
  {
    const __bf16* qrow = Q + (size_t)((q0 + m16) * 1024 + h * 64);
    for (int kh = 0; kh < 2; ++kh) {
      qa[kh].f[0] = *(const float4*)(qrow + kh * 32 + lh * 8);
      qa[kh].f[1] = *(const float4*)(qrow + kh * 32 + lh * 8 + 16);
    }
  }

  float m_s[8], l_s[8];
  v8f acc[4];
  for (int r = 0; r < 8; ++r) { m_s[r] = -1e30f; l_s[r] = 0.f; }
  for (int c = 0; c < 4; ++c)
    for (int r = 0; r < 8; ++r) acc[c][r] = 0.f;

  const float sc = 0.125f;  // 1/sqrt(64)

  for (int j0 = 0; j0 < q0 + 16; j0 += 32) {
    // S = Q @ K^T for keys [j0, j0+32): two 16x16 f32 tiles.
    v8f sv[2];
    for (int t = 0; t < 2; ++t) {
      v8f s = {0.f, 0.f, 0.f, 0.f, 0.f, 0.f, 0.f, 0.f};
      for (int kh = 0; kh < 2; ++kh) {
        Frag kf;  // B fragment: lane m16 = key column, elements contiguous in d.
        const __bf16* krow =
            Km + (size_t)((j0 + t * 16 + m16) * 1024 + h * 64 + kh * 32 + lh * 8);
        kf.f[0] = *(const float4*)(krow);
        kf.f[1] = *(const float4*)(krow + 16);
        s = wmma_bf16(qa[kh].v, kf.v, s);
      }
      sv[t] = s;
    }

    // Online softmax; row stats reduced across 16-lane groups (rows 0-7 in lanes 0-15,
    // rows 8-15 in lanes 16-31, matching the f32 C layout).
    for (int r = 0; r < 8; ++r) {
      int row  = q0 + lh * 8 + r;
      int key0 = j0 + m16;
      float s0 = (key0      <= row) ? sv[0][r] * sc : -1e30f;
      float s1 = (key0 + 16 <= row) ? sv[1][r] * sc : -1e30f;
      float mx = fmaxf(s0, s1);
      mx = fmaxf(mx, __shfl_xor(mx, 1, 32));
      mx = fmaxf(mx, __shfl_xor(mx, 2, 32));
      mx = fmaxf(mx, __shfl_xor(mx, 4, 32));
      mx = fmaxf(mx, __shfl_xor(mx, 8, 32));
      float mn = fmaxf(m_s[r], mx);
      float al = __expf(m_s[r] - mn);
      float p0 = __expf(s0 - mn);
      float p1 = __expf(s1 - mn);
      float ps = p0 + p1;
      ps += __shfl_xor(ps, 1, 32);
      ps += __shfl_xor(ps, 2, 32);
      ps += __shfl_xor(ps, 4, 32);
      ps += __shfl_xor(ps, 8, 32);
      l_s[r] = l_s[r] * al + ps;
      m_s[r] = mn;
      for (int c = 0; c < 4; ++c) acc[c][r] *= al;
      Pw[(lh * 8 + r) * 40 +      m16] = (__bf16)p0;
      Pw[(lh * 8 + r) * 40 + 16 + m16] = (__bf16)p1;
    }
    DS_WAIT();  // P stores visible before re-reading in A-fragment layout

    Frag pf;  // P as 16x32 A fragment
    {
      const __bf16* prow = Pw + m16 * 40 + lh * 8;
      pf.f[0] = *(const float4*)(prow);
      pf.f[1] = *(const float4*)(prow + 16);
    }
    // ctx += P @ V_block; V B fragments contiguous thanks to pre-transposed Vt.
    for (int c = 0; c < 4; ++c) {
      Frag vf;
      const __bf16* vrow = Vt + (size_t)((h * 64 + c * 16 + m16) * 4096 + j0 + lh * 8);
      vf.f[0] = *(const float4*)(vrow);
      vf.f[1] = *(const float4*)(vrow + 16);
      acc[c] = wmma_bf16(pf.v, vf.v, acc[c]);
    }
  }

  for (int c = 0; c < 4; ++c) {
    __bf16* p = Ctx + (size_t)((q0 + lh * 8) * 1024 + h * 64 + c * 16 + m16);
    for (int r = 0; r < 8; ++r)
      p[r * 1024] = (__bf16)(acc[c][r] / l_s[r]);
  }
}

extern "C" void kernel_launch(void* const* d_in, const int* in_sizes, int n_in,
                              void* d_out, int out_size, void* d_ws, size_t ws_size,
                              hipStream_t stream) {
  (void)in_sizes; (void)n_in; (void)out_size; (void)ws_size;
  const float* x    = (const float*)d_in[0];
  const float* Wdkv = (const float*)d_in[1];
  const float* Wuk  = (const float*)d_in[2];
  const float* Wuv  = (const float*)d_in[3];
  const float* Wuq  = (const float*)d_in[4];
  const float* Wo   = (const float*)d_in[5];
  float* out = (float*)d_out;

  const int T = 4096, D = 1024, L = 256;

  char* ws = (char*)d_ws;
  size_t off = 0;
  auto alloc = [&](size_t elems) -> __bf16* {
    __bf16* p = (__bf16*)(ws + off);
    off += ((elems * sizeof(__bf16)) + 255) & ~(size_t)255;
    return p;
  };
  __bf16* xb   = alloc((size_t)T * D);
  __bf16* wdkv = alloc((size_t)D * L);
  __bf16* wuk  = alloc((size_t)L * D);
  __bf16* wuv  = alloc((size_t)L * D);
  __bf16* wuq  = alloc((size_t)D * D);
  __bf16* wob  = alloc((size_t)D * D);
  __bf16* ckv  = alloc((size_t)T * L);
  __bf16* qb   = alloc((size_t)T * D);
  __bf16* kb   = alloc((size_t)T * D);
  __bf16* vt   = alloc((size_t)T * D);
  __bf16* ctx  = alloc((size_t)T * D);

  auto cvt = [&](const float* s, __bf16* d, int n) {
    cvt_f32_bf16_kernel<<<(n + 255) / 256, 256, 0, stream>>>(s, d, n);
  };
  cvt(x,    xb,   T * D);
  cvt(Wdkv, wdkv, D * L);
  cvt(Wuk,  wuk,  L * D);
  cvt(Wuv,  wuv,  L * D);
  cvt(Wuq,  wuq,  D * D);
  cvt(Wo,   wob,  D * D);

  // c_kv = x @ W_DKV : (T, L)
  gemm_bf16_wmma<<<dim3(L / 128, T / 128), 256, 0, stream>>>(xb, wdkv, ckv, T, L, D, D, L, L, 0);
  // q = x @ W_UQ : (T, D)
  gemm_bf16_wmma<<<dim3(D / 128, T / 128), 256, 0, stream>>>(xb, wuq, qb, T, D, D, D, D, D, 0);
  // k = c_kv @ W_UK : (T, D)
  gemm_bf16_wmma<<<dim3(D / 128, T / 128), 256, 0, stream>>>(ckv, wuk, kb, T, D, L, L, D, D, 0);
  // v^T = (c_kv @ W_UV)^T : stored (D, T) for contiguous V B-fragments
  gemm_bf16_wmma<<<dim3(D / 128, T / 128), 256, 0, stream>>>(ckv, wuv, vt, T, D, L, L, D, T, 1);
  // causal flash attention -> ctx (T, D)
  mla_flash_attn<<<dim3(16, T / 64), 128, 0, stream>>>(qb, kb, vt, ctx);
  // out = ctx @ W_O : f32
  gemm_bf16_wmma<<<dim3(D / 128, T / 128), 256, 0, stream>>>(ctx, wob, out, T, D, D, D, D, D, 2);
}
